// MaskBBoxPredictor_19086834663565
// MI455X (gfx1250) — matrix operand order repaired and checked by reading, not compiled, not run on hardware
//
#include <hip/hip_runtime.h>
#include <hip/hip_bf16.h>

typedef _Float16 half_t;
typedef __attribute__((ext_vector_type(8)))  _Float16 v8h;
typedef __attribute__((ext_vector_type(16))) _Float16 v16h;
typedef __attribute__((ext_vector_type(8)))  float    v8f;

#define NSEG 33
#define EPS_GN 1e-5f

// ---------------------------------------------------------------------------
// Weight repack: [Cout][Cin][3][3] f32 -> [tap][Cin/32][Cout][32] f16.
// K-blocked and k-contiguous per cout so the conv kernel's B stage is a
// straight (vectorized) global->LDS copy with zero transposition work.
// ---------------------------------------------------------------------------
__global__ void k_pack_w(const float* __restrict__ w, half_t* __restrict__ wp,
                         int Cout, int Cin) {
  int idx = blockIdx.x * blockDim.x + threadIdx.x;
  int total = Cout * Cin * 9;
  if (idx >= total) return;
  int tap = idx % 9;
  int ci  = (idx / 9) % Cin;
  int co  = idx / (9 * Cin);
  int kb = ci >> 5, kk = ci & 31;
  wp[(((long long)tap * (Cin >> 5) + kb) * Cout + co) * 32 + kk] = (half_t)w[idx];
}

// ---------------------------------------------------------------------------
// Build layer-0 activations: NHWC f16, C = 512 (from x, NCHW f32) + 64 coord
// coord features: relu(wc[c][0]*gx + wc[c][1]*gy + bc[c])
// ---------------------------------------------------------------------------
__global__ void k_build_act0(const float* __restrict__ x,
                             const float* __restrict__ wc,
                             const float* __restrict__ bc,
                             half_t* __restrict__ act0,
                             int B, int H, int W, int Cf) {
  int C = Cf + 64;
  long long total = (long long)B * C * H * W;
  long long idx = (long long)blockIdx.x * blockDim.x + threadIdx.x;
  if (idx >= total) return;
  int wi = (int)(idx % W);
  int hi = (int)((idx / W) % H);
  int c  = (int)((idx / ((long long)W * H)) % C);
  int b  = (int)(idx / ((long long)W * H * C));
  float v;
  if (c < Cf) {
    v = x[(((long long)b * Cf + c) * H + hi) * W + wi];
  } else {
    int cc = c - Cf;
    v = wc[cc * 2 + 0] * (float)wi + wc[cc * 2 + 1] * (float)hi + bc[cc];
    v = v > 0.f ? v : 0.f;
  }
  act0[(((long long)b * H + hi) * W + wi) * C + c] = (half_t)v;
}

// ---------------------------------------------------------------------------
// Implicit-GEMM conv3x3 (pad=1) with v_wmma_f32_16x16x32_f16.
// in : NHWC f16 [B,H,W,Cin]    wp : [tap][Cin/32][Cout][32] f16
// out: NHWC f32 [B,H,W,Cout]   (conv + bias; GN applied later)
// Block: 256 threads (8 waves). Tile: 128 pixels (one full W row) x 64 couts.
// Wave (wm,wn) computes 32 pixels x 32 couts: 4 accumulators, 4 WMMAs/K-step.
// ---------------------------------------------------------------------------
__global__ __launch_bounds__(256) void k_conv3x3_wmma(
    const half_t* __restrict__ in, const half_t* __restrict__ wp,
    const float* __restrict__ bias, float* __restrict__ out,
    int B, int H, int W, int Cin, int Cout) {
  const int LDA = 40;  // halfs per LDS row: 80B, 16B-aligned, bank-skewed
  __shared__ __align__(16) half_t ldsA[128 * 40];  // [pixel][k]
  __shared__ __align__(16) half_t ldsBt[64 * 40];  // [cout][k]

  int t    = threadIdx.x;
  int lane = t & 31;
  int wave = t >> 5;
  int wm = wave & 3, wn = wave >> 2;
  int lrow = lane & 15;
  int hi16 = (lane >> 4) & 1;

  int h  = blockIdx.x % H;
  int b  = blockIdx.x / H;
  int n0 = blockIdx.y * 64;
  int KB = Cin >> 5;

  v8f c00 = {}, c01 = {}, c10 = {}, c11 = {};

  for (int tap = 0; tap < 9; ++tap) {
    int dh = tap / 3 - 1, dw = tap % 3 - 1;
    int hin = h + dh;
    if (hin < 0 || hin >= H) continue;  // uniform branch (h uniform per block)
    const half_t* inRow = in + (long long)(b * H + hin) * W * Cin;
    const half_t* wTap  = wp + (long long)tap * KB * Cout * 32;

    for (int kb = 0; kb < KB; ++kb) {
      // --- stage A: 128 pixels x 32 ch (two 16B loads/thread, border zfill) ---
      {
        int i0 = t >> 2;
        int cc = (t & 3) * 8;
        #pragma unroll
        for (int rep = 0; rep < 2; ++rep) {
          int i = i0 + rep * 64;
          int win = i + dw;
          v8h v = {};
          if (win >= 0 && win < W)
            v = *(const v8h*)(inRow + (long long)win * Cin + kb * 32 + cc);
          *(v8h*)&ldsA[i * LDA + cc] = v;
        }
      }
      // --- stage B: 64 couts x 32 k, already k-major: straight b128 copy ---
      {
        int n  = t >> 2;
        int kk = (t & 3) * 8;
        const half_t* src = wTap + ((long long)kb * Cout + n0 + n) * 32 + kk;
        *(v8h*)&ldsBt[n * LDA + kk] = *(const v8h*)src;
        if (kb + 1 < KB)
          __builtin_prefetch(src + (long long)Cout * 32, 0, 1);
      }
      __syncthreads();

      // --- per-lane fragment gather (ISA 16-bit A/B VGPR striping) ---
      v16h a0, a1, b0, b1;
      {
        int o = hi16 * 8;  // lanes 16..31 take K halves [8..15]/[24..31]
        int r0 = wm * 32 + lrow;
        v8h lo0 = *(const v8h*)&ldsA[r0 * LDA + o];
        v8h hv0 = *(const v8h*)&ldsA[r0 * LDA + 16 + o];
        int r1 = r0 + 16;
        v8h lo1 = *(const v8h*)&ldsA[r1 * LDA + o];
        v8h hv1 = *(const v8h*)&ldsA[r1 * LDA + 16 + o];
        #pragma unroll
        for (int e = 0; e < 8; ++e) {
          a0[e] = lo0[e]; a0[e + 8] = hv0[e];
          a1[e] = lo1[e]; a1[e + 8] = hv1[e];
        }
      }
      {
        int kk  = hi16 * 8;  // lanes 16..31 take K halves [8..15]/[24..31]
        int nc0 = wn * 32 + lrow;
        v8h lo0 = *(const v8h*)&ldsBt[nc0 * LDA + kk];
        v8h hv0 = *(const v8h*)&ldsBt[nc0 * LDA + kk + 16];
        int nc1 = nc0 + 16;
        v8h lo1 = *(const v8h*)&ldsBt[nc1 * LDA + kk];
        v8h hv1 = *(const v8h*)&ldsBt[nc1 * LDA + kk + 16];
        #pragma unroll
        for (int e = 0; e < 8; ++e) {
          b0[e] = lo0[e]; b0[e + 8] = hv0[e];
          b1[e] = lo1[e]; b1[e + 8] = hv1[e];
        }
      }
      c00 = __builtin_amdgcn_wmma_f32_16x16x32_f16(false, a0, false, b0, (short)0, c00, false, false);
      c01 = __builtin_amdgcn_wmma_f32_16x16x32_f16(false, a0, false, b1, (short)0, c01, false, false);
      c10 = __builtin_amdgcn_wmma_f32_16x16x32_f16(false, a1, false, b0, (short)0, c10, false, false);
      c11 = __builtin_amdgcn_wmma_f32_16x16x32_f16(false, a1, false, b1, (short)0, c11, false, false);
      __syncthreads();
    }
  }

  // --- epilogue: C layout lanes0-15 -> M=r, lanes16-31 -> M=r+8 ---
  long long pbase = (long long)(b * H + h) * W;
  int col0 = n0 + wn * 32 + lrow;
  float bs0 = bias[col0];
  float bs1 = bias[col0 + 16];
  #pragma unroll
  for (int r = 0; r < 8; ++r) {
    int m0 = wm * 32 + r + hi16 * 8;
    long long off0 = (pbase + m0) * Cout;
    out[off0 + col0]      = c00[r] + bs0;
    out[off0 + col0 + 16] = c01[r] + bs1;
    long long off1 = (pbase + m0 + 16) * Cout;
    out[off1 + col0]      = c10[r] + bs0;
    out[off1 + col0 + 16] = c11[r] + bs1;
  }
}

// ---------------------------------------------------------------------------
// GroupNorm pass 1: per (b, group) mean / rstd over H*W*(C/G). NHWC f32 input.
// ---------------------------------------------------------------------------
__global__ __launch_bounds__(256) void k_gn_stats(const float* __restrict__ y,
                                                  float* __restrict__ stats,
                                                  int HW, int C, int G) {
  int bg = blockIdx.x;
  int g = bg % G, b = bg / G;
  int Cg = C / G;
  long long n = (long long)HW * Cg;
  const float* base = y + (long long)b * HW * C + g * Cg;
  float s1 = 0.f, s2 = 0.f;
  for (long long i = threadIdx.x; i < n; i += blockDim.x) {
    int cg = (int)(i % Cg);
    long long pix = i / Cg;
    float v = base[pix * C + cg];
    s1 += v; s2 += v * v;
  }
  __shared__ float r1[256], r2[256];
  r1[threadIdx.x] = s1; r2[threadIdx.x] = s2;
  __syncthreads();
  for (int s = 128; s > 0; s >>= 1) {
    if ((int)threadIdx.x < s) {
      r1[threadIdx.x] += r1[threadIdx.x + s];
      r2[threadIdx.x] += r2[threadIdx.x + s];
    }
    __syncthreads();
  }
  if (threadIdx.x == 0) {
    float mean = r1[0] / (float)n;
    float var  = r2[0] / (float)n - mean * mean;
    stats[bg * 2 + 0] = mean;
    stats[bg * 2 + 1] = rsqrtf(var + EPS_GN);
  }
}

// ---------------------------------------------------------------------------
// GroupNorm pass 2 + ReLU, f32 NHWC -> f16 NHWC (next conv input / pooling)
// ---------------------------------------------------------------------------
__global__ void k_gn_apply(const float* __restrict__ y, const float* __restrict__ stats,
                           const float* __restrict__ gamma, const float* __restrict__ beta,
                           half_t* __restrict__ act, long long total, int HW, int C, int G) {
  long long idx = (long long)blockIdx.x * blockDim.x + threadIdx.x;
  if (idx >= total) return;
  int c = (int)(idx % C);
  int b = (int)(idx / ((long long)HW * C));
  int g = c / (C / G);
  float mean = stats[(b * G + g) * 2 + 0];
  float rstd = stats[(b * G + g) * 2 + 1];
  float v = (y[idx] - mean) * rstd * gamma[c] + beta[c];
  act[idx] = (half_t)(v > 0.f ? v : 0.f);
}

__global__ void k_zero(float* p, int n) {
  int i = blockIdx.x * blockDim.x + threadIdx.x;
  if (i < n) p[i] = 0.f;
}

// ---------------------------------------------------------------------------
// Masked segment pooling: LDS-accumulated partial sums, global atomic flush.
// act: NHWC f16 [B,H,W,64]
// ---------------------------------------------------------------------------
__global__ __launch_bounds__(256) void k_pool(const half_t* __restrict__ act,
                                              const int* __restrict__ masks,
                                              float* __restrict__ sums,
                                              float* __restrict__ cnts, int HW) {
  __shared__ float ps[NSEG * 64];
  __shared__ float pc[NSEG];
  int t = threadIdx.x;
  for (int i = t; i < NSEG * 64; i += 256) ps[i] = 0.f;
  if (t < NSEG) pc[t] = 0.f;
  __syncthreads();

  int b = blockIdx.y;
  int pix = blockIdx.x * 256 + t;
  int m = masks[(long long)b * HW + pix];
  const half_t* a = act + ((long long)b * HW + pix) * 64;
  atomicAdd(&pc[m], 1.f);
  #pragma unroll 8
  for (int c = 0; c < 64; ++c) atomicAdd(&ps[m * 64 + c], (float)a[c]);
  __syncthreads();

  float* gs = sums + (long long)b * NSEG * 64;
  for (int i = t; i < NSEG * 64; i += 256)
    if (ps[i] != 0.f) atomicAdd(&gs[i], ps[i]);
  if (t < NSEG && pc[t] != 0.f) atomicAdd(&cnts[b * NSEG + t], pc[t]);
}

// ---------------------------------------------------------------------------
// Heads: pooled mean (safe divide), then 64->7 box and 64->1 conf projections.
// out layout: boxes [B,32,7] then scores [B,32]
// ---------------------------------------------------------------------------
__global__ void k_heads(const float* __restrict__ sums, const float* __restrict__ cnts,
                        const float* __restrict__ wbox, const float* __restrict__ bbox,
                        const float* __restrict__ wconf, const float* __restrict__ bconf,
                        float* __restrict__ outp, int Bn) {
  int t = blockIdx.x * blockDim.x + threadIdx.x;
  if (t >= Bn * 32) return;
  int b = t / 32, o = t % 32;
  int seg = b * NSEG + o + 1;  // drop background segment 0
  float inv = 1.f / fmaxf(cnts[seg], 1e-4f);
  float p[64];
  #pragma unroll
  for (int c = 0; c < 64; ++c) p[c] = sums[(long long)seg * 64 + c] * inv;
  for (int k = 0; k < 7; ++k) {
    float acc = bbox[k];
    for (int c = 0; c < 64; ++c) acc += p[c] * wbox[k * 64 + c];
    outp[(b * 32 + o) * 7 + k] = acc;
  }
  float acc = bconf[0];
  for (int c = 0; c < 64; ++c) acc += p[c] * wconf[c];
  outp[Bn * 32 * 7 + b * 32 + o] = acc;
}

// ---------------------------------------------------------------------------
extern "C" void kernel_launch(void* const* d_in, const int* in_sizes, int n_in,
                              void* d_out, int out_size, void* d_ws, size_t ws_size,
                              hipStream_t stream) {
  const int B = 4, H = 128, W = 128, Cf = 512, C0 = Cf + 64;
  const int HW = H * W;

  const float* x     = (const float*)d_in[0];
  const int*   masks = (const int*)d_in[1];
  const float* wc    = (const float*)d_in[2];
  const float* bcd   = (const float*)d_in[3];
  const float* w1 = (const float*)d_in[4];  const float* b1 = (const float*)d_in[5];
  const float* g1 = (const float*)d_in[6];  const float* t1 = (const float*)d_in[7];
  const float* w2 = (const float*)d_in[8];  const float* b2 = (const float*)d_in[9];
  const float* g2 = (const float*)d_in[10]; const float* t2 = (const float*)d_in[11];
  const float* w3 = (const float*)d_in[12]; const float* b3 = (const float*)d_in[13];
  const float* g3 = (const float*)d_in[14]; const float* t3 = (const float*)d_in[15];
  const float* wbox  = (const float*)d_in[16];
  const float* bbox  = (const float*)d_in[17];
  const float* wconf = (const float*)d_in[18];
  const float* bconf = (const float*)d_in[19];
  float* outp = (float*)d_out;

  // ---- workspace layout (two reused arenas + packed weights + small bufs) ----
  char* ws = (char*)d_ws;
  size_t off = 0;
  auto carve = [&](size_t bytes) -> char* {
    off = (off + 255) & ~(size_t)255;
    char* p = ws + off;
    off += bytes;
    return p;
  };
  half_t* arenaA = (half_t*)carve((size_t)B * HW * C0 * 2);   // act0 / act1 / act2 / act3
  float*  arenaB = (float*) carve((size_t)B * HW * 256 * 4);  // y1 / y2 / y3
  half_t* wp1 = (half_t*)carve((size_t)9 * C0  * 256 * 2);
  half_t* wp2 = (half_t*)carve((size_t)9 * 256 * 128 * 2);
  half_t* wp3 = (half_t*)carve((size_t)9 * 128 * 64  * 2);
  float*  stats = (float*)carve(64 * 4);                      // up to 32 (b,g) * {mu,rstd}
  float*  pooled = (float*)carve((size_t)(B * NSEG * 64 + B * NSEG) * 4);
  float*  pcnts  = pooled + B * NSEG * 64;

  // 1) pack weights -> [tap][Cin/32][Cout][32] f16
  {
    int n1 = 256 * C0 * 9, n2 = 128 * 256 * 9, n3 = 64 * 128 * 9;
    k_pack_w<<<(n1 + 255) / 256, 256, 0, stream>>>(w1, wp1, 256, C0);
    k_pack_w<<<(n2 + 255) / 256, 256, 0, stream>>>(w2, wp2, 128, 256);
    k_pack_w<<<(n3 + 255) / 256, 256, 0, stream>>>(w3, wp3, 64, 128);
  }
  // 2) layer-0 activations (x NCHW f32 + coord features) -> NHWC f16
  {
    long long n = (long long)B * C0 * HW;
    k_build_act0<<<(unsigned)((n + 255) / 256), 256, 0, stream>>>(x, wc, bcd, arenaA, B, H, W, Cf);
  }
  const int mtiles = B * H;  // one full 128-pixel row per block tile

  // 3) conv1 (576 -> 256) + GN(8) + ReLU
  k_conv3x3_wmma<<<dim3(mtiles, 256 / 64), 256, 0, stream>>>(
      arenaA, wp1, b1, arenaB, B, H, W, C0, 256);
  k_gn_stats<<<B * 8, 256, 0, stream>>>(arenaB, stats, HW, 256, 8);
  {
    long long n = (long long)B * HW * 256;
    k_gn_apply<<<(unsigned)((n + 255) / 256), 256, 0, stream>>>(
        arenaB, stats, g1, t1, arenaA, n, HW, 256, 8);
  }
  // 4) conv2 (256 -> 128) + GN(4) + ReLU
  k_conv3x3_wmma<<<dim3(mtiles, 128 / 64), 256, 0, stream>>>(
      arenaA, wp2, b2, arenaB, B, H, W, 256, 128);
  k_gn_stats<<<B * 4, 256, 0, stream>>>(arenaB, stats, HW, 128, 4);
  {
    long long n = (long long)B * HW * 128;
    k_gn_apply<<<(unsigned)((n + 255) / 256), 256, 0, stream>>>(
        arenaB, stats, g2, t2, arenaA, n, HW, 128, 4);
  }
  // 5) conv3 (128 -> 64) + GN(4) + ReLU
  k_conv3x3_wmma<<<dim3(mtiles, 64 / 64), 256, 0, stream>>>(
      arenaA, wp3, b3, arenaB, B, H, W, 128, 64);
  k_gn_stats<<<B * 4, 256, 0, stream>>>(arenaB, stats, HW, 64, 4);
  {
    long long n = (long long)B * HW * 64;
    k_gn_apply<<<(unsigned)((n + 255) / 256), 256, 0, stream>>>(
        arenaB, stats, g3, t3, arenaA, n, HW, 64, 4);
  }
  // 6) masked segment pooling (zeroed every launch for determinism)
  {
    int nz = B * NSEG * 64 + B * NSEG;
    k_zero<<<(nz + 255) / 256, 256, 0, stream>>>(pooled, nz);
    k_pool<<<dim3(HW / 256, B), 256, 0, stream>>>(arenaA, masks, pooled, pcnts, HW);
  }
  // 7) heads
  k_heads<<<1, 128, 0, stream>>>(pooled, pcnts, wbox, bbox, wconf, bconf, outp, B);
}
